// GAT_Pruning_20916490731916
// MI455X (gfx1250) — matrix operand order repaired and checked by reading, not compiled, not run on hardware
//
#include <hip/hip_runtime.h>
#include <hip/hip_bf16.h>

// ---- problem constants (match reference) ----
#define HID     512
#define NH      8
#define CH      64
#define IN_F    256
#define OUT_C   40
#define HHID    256
#define BN_EPS  1e-5f
#define SM_EPS  1e-16f

typedef __attribute__((ext_vector_type(16))) __bf16 v16bf;
typedef __attribute__((ext_vector_type(8)))  float  v8f;

// ---- monotone float<->uint key (for atomic max / radix select) ----
__device__ __forceinline__ unsigned orderKey(float f) {
  unsigned b = __float_as_uint(f);
  return (b & 0x80000000u) ? ~b : (b | 0x80000000u);
}
__device__ __forceinline__ float keyToFloat(unsigned k) {
  unsigned b = (k & 0x80000000u) ? (k & 0x7fffffffu) : ~k;
  return __uint_as_float(b);
}

// ---- CDNA5 async global->LDS copy (ASYNCcnt-tracked), 16B per lane ----
__device__ __forceinline__ void async_copy16(unsigned lds_off, const void* g) {
  unsigned long long ga = (unsigned long long)(uintptr_t)g;
  asm volatile("global_load_async_to_lds_b128 %0, %1, off"
               :: "v"(lds_off), "v"(ga)
               : "memory");
}
__device__ __forceinline__ void wait_async0() {
  asm volatile("s_wait_asynccnt 0x0" ::: "memory");
}

// ======================================================================
// Weight packing: B (K x Nc, f32 row-major) -> bf16 WMMA B-fragments.
// Contiguous-k layout, not ISA table order (16-bit B 32x16): lane = n (0..15) | half*16,
// lanes 0-15 hold K=k0..k0+15 (2 per VGPR), lanes 16-31 hold K=k0+16..k0+31.
// Packed as [kt][nt][lane][16 bf16] => each lane's fragment is 32 contiguous
// bytes, and a 4-n-tile panel for one k-step is one contiguous 4KB region.
// ======================================================================
__global__ void pack_b_k(const float* __restrict__ B, __bf16* __restrict__ Bp,
                         int K, int Nc) {
  int idx = blockIdx.x * blockDim.x + threadIdx.x;
  int ntiles = Nc >> 4, ktiles = K >> 5;
  int total = ktiles * ntiles * 32;
  if (idx >= total) return;
  int lane = idx & 31;
  int t = idx >> 5;
  int nt = t % ntiles, kt = t / ntiles;
  int n = (nt << 4) + (lane & 15);
  int kbase = (kt << 5) + ((lane >> 4) << 4);
  __bf16* out = Bp + ((size_t)idx << 4);
#pragma unroll
  for (int e = 0; e < 16; ++e)
    out[e] = (__bf16)B[(size_t)(kbase + e) * Nc + n];
}

// ======================================================================
// GEMM: C[M x Nc] = A[M x K] @ B + bias, bf16 WMMA, f32 accumulate.
// Block = 8 waves: wave w owns m-tile (blockIdx.x*8+w) and a 16x64 output
// strip (4 accumulators over n-tiles nb*4..nb*4+3). The shared 4KB B panel
// per k-step is staged into LDS with async copies, double-buffered.
// ======================================================================
__global__ __launch_bounds__(256) void gemm_wmma_k(
    const float* __restrict__ A, const __bf16* __restrict__ Bp,
    const float* __restrict__ bias, float* __restrict__ C,
    int M, int K, int Nc) {
  __shared__ __align__(32) char sbuf[2][4096];
  const int t = threadIdx.x;
  const int lane = t & 31;
  const int wave = t >> 5;
  const int ntiles = Nc >> 4;
  const int mt = blockIdx.x * 8 + wave;
  const int nb4 = blockIdx.y << 2;
  const int half = lane >> 4;
  const int m = (mt << 4) + (lane & 15);
  const int mrow = (m < M) ? m : (M - 1);
  const float* Ar = A + (size_t)mrow * K;
  const int ktiles = K >> 5;
  const __bf16* Bblk = Bp + (((size_t)nb4 * 32) << 4);   // + kt*ntiles*32*16
  const size_t bkstride = ((size_t)ntiles * 32) << 4;    // bf16 elems per k-step
  const unsigned ldsbase = (unsigned)(uintptr_t)&sbuf[0][0];

  v8f acc0 = {0.f,0.f,0.f,0.f,0.f,0.f,0.f,0.f};
  v8f acc1 = acc0, acc2 = acc0, acc3 = acc0;

  // prologue: stage k-step 0 (all 256 threads copy 16B each = 4KB)
  async_copy16(ldsbase + t * 16, (const char*)Bblk + (size_t)t * 16);
  wait_async0();
  __syncthreads();

  int cur = 0;
  for (int kt = 0; kt < ktiles; ++kt) {
    if (kt + 1 < ktiles) {  // prefetch next B panel into the other buffer
      const char* src = (const char*)(Bblk + (size_t)(kt + 1) * bkstride) + (size_t)t * 16;
      async_copy16(ldsbase + (unsigned)((cur ^ 1) * 4096 + t * 16), src);
    }
    // A fragment: one contiguous 16-K chunk per half (matches packed B order)
    const float* p = Ar + (kt << 5) + half * 16;
    float4 f0 = *(const float4*)(p);
    float4 f1 = *(const float4*)(p + 4);
    float4 f2 = *(const float4*)(p + 8);
    float4 f3 = *(const float4*)(p + 12);
    v16bf a;
    a[0]  = (__bf16)f0.x; a[1]  = (__bf16)f0.y; a[2]  = (__bf16)f0.z; a[3]  = (__bf16)f0.w;
    a[4]  = (__bf16)f1.x; a[5]  = (__bf16)f1.y; a[6]  = (__bf16)f1.z; a[7]  = (__bf16)f1.w;
    a[8]  = (__bf16)f2.x; a[9]  = (__bf16)f2.y; a[10] = (__bf16)f2.z; a[11] = (__bf16)f2.w;
    a[12] = (__bf16)f3.x; a[13] = (__bf16)f3.y; a[14] = (__bf16)f3.z; a[15] = (__bf16)f3.w;

    const char* bb = &sbuf[cur][0];
    v16bf b0 = *(const v16bf*)(bb + ((0 * 32 + lane) << 5));
    acc0 = __builtin_amdgcn_wmma_f32_16x16x32_bf16(false, a, false, b0, (short)0, acc0, false, false);
    v16bf b1 = *(const v16bf*)(bb + ((1 * 32 + lane) << 5));
    acc1 = __builtin_amdgcn_wmma_f32_16x16x32_bf16(false, a, false, b1, (short)0, acc1, false, false);
    v16bf b2 = *(const v16bf*)(bb + ((2 * 32 + lane) << 5));
    acc2 = __builtin_amdgcn_wmma_f32_16x16x32_bf16(false, a, false, b2, (short)0, acc2, false, false);
    v16bf b3 = *(const v16bf*)(bb + ((3 * 32 + lane) << 5));
    acc3 = __builtin_amdgcn_wmma_f32_16x16x32_bf16(false, a, false, b3, (short)0, acc3, false, false);

    wait_async0();      // own async copies for next buffer complete
    __syncthreads();    // whole block's next buffer visible; cur free to reuse
    cur ^= 1;
  }

  const int rbase = (mt << 4) + half * 8;
  if ((mt << 4) < M) {
#define STORE_TILE(ACC, NTL)                                                  \
    {                                                                         \
      int n = ((nb4 + (NTL)) << 4) + (lane & 15);                             \
      float bv = bias ? bias[n] : 0.f;                                        \
      _Pragma("unroll")                                                       \
      for (int r = 0; r < 8; ++r) {                                           \
        int row = rbase + r;                                                  \
        if (row < M) C[(size_t)row * Nc + n] = ACC[r] + bv;                   \
      }                                                                       \
    }
    STORE_TILE(acc0, 0)
    STORE_TILE(acc1, 1)
    STORE_TILE(acc2, 2)
    STORE_TILE(acc3, 3)
#undef STORE_TILE
  }
}

// ======================================================================
// Per-node attention coefficients: a_src/a_dst[n,h] = dot(xp[n,h,:], a[h,:])
// ======================================================================
__global__ void attn_coeff_k(const float* __restrict__ xp, const float* __restrict__ asrc,
                             const float* __restrict__ adst, float* __restrict__ as_,
                             float* __restrict__ ad_, int n_nodes) {
  int i = blockIdx.x * blockDim.x + threadIdx.x;
  if (i >= n_nodes * NH) return;
  int n = i / NH, h = i % NH;
  const float* xr = xp + (size_t)n * HID + h * CH;
  const float* s1 = asrc + h * CH;
  const float* s2 = adst + h * CH;
  float s = 0.f, d = 0.f;
#pragma unroll 8
  for (int c = 0; c < CH; ++c) { float v = xr[c]; s += v * s1[c]; d += v * s2[c]; }
  as_[i] = s; ad_[i] = d;
}

// edge id e: e < ne -> (srcA[e], dstA[e]); else self-loop (e-ne, e-ne)
__global__ void edge_logits_k(const int* __restrict__ srcA, const int* __restrict__ dstA,
                              int ne, int n_nodes, const float* __restrict__ as_,
                              const float* __restrict__ ad_, float* __restrict__ elog,
                              unsigned* __restrict__ mx) {
  int idx = blockIdx.x * blockDim.x + threadIdx.x;
  int total = (ne + n_nodes) * NH;
  if (idx >= total) return;
  int e = idx / NH, h = idx % NH;
  int s = (e < ne) ? srcA[e] : (e - ne);
  int d = (e < ne) ? dstA[e] : (e - ne);
  float v = as_[s * NH + h] + ad_[d * NH + h];
  v = (v > 0.f) ? v : 0.2f * v;          // leaky_relu(0.2)
  elog[idx] = v;
  atomicMax(&mx[d * NH + h], orderKey(v));
}

__global__ void edge_exp_k(const int* __restrict__ dstA, int ne, int n_nodes,
                           float* __restrict__ elog, const unsigned* __restrict__ mx,
                           float* __restrict__ den) {
  int idx = blockIdx.x * blockDim.x + threadIdx.x;
  int total = (ne + n_nodes) * NH;
  if (idx >= total) return;
  int e = idx / NH, h = idx % NH;
  int d = (e < ne) ? dstA[e] : (e - ne);
  float ex = __expf(elog[idx] - keyToFloat(mx[d * NH + h]));
  elog[idx] = ex;
  atomicAdd(&den[d * NH + h], ex);
}

__global__ void edge_alpha_k(const int* __restrict__ dstA, int ne, int n_nodes,
                             float* __restrict__ elog, const float* __restrict__ den) {
  int idx = blockIdx.x * blockDim.x + threadIdx.x;
  int total = (ne + n_nodes) * NH;
  if (idx >= total) return;
  int e = idx / NH, h = idx % NH;
  int d = (e < ne) ? dstA[e] : (e - ne);
  elog[idx] = elog[idx] / (den[d * NH + h] + SM_EPS);
}

__global__ void edge_score_k(const float* __restrict__ alpha, float* __restrict__ score,
                             int total) {
  int e = blockIdx.x * blockDim.x + threadIdx.x;
  if (e >= total) return;
  float s = 0.f;
#pragma unroll
  for (int h = 0; h < NH; ++h) s += alpha[(size_t)e * NH + h];
  score[e] = s * (1.f / NH);
}

// out[dst,f] += xp[src,f] * alpha[e, f/64]   (thread per edge-feature)
__global__ void aggregate_k(const int* __restrict__ srcA, const int* __restrict__ dstA,
                            int ne, int n_nodes, const float* __restrict__ xp,
                            const float* __restrict__ alpha, float* __restrict__ outp) {
  size_t idx = (size_t)blockIdx.x * blockDim.x + threadIdx.x;
  size_t total = (size_t)(ne + n_nodes) * HID;
  if (idx >= total) return;
  int f = (int)(idx & (HID - 1));
  int e = (int)(idx >> 9);
  int s = (e < ne) ? srcA[e] : (e - ne);
  int d = (e < ne) ? dstA[e] : (e - ne);
  float al = alpha[(size_t)e * NH + (f >> 6)];
  atomicAdd(&outp[(size_t)d * HID + f], xp[(size_t)s * HID + f] * al);
}

// h = elu(bn(h + bias)) [+ residual]
__global__ void bias_bn_elu_k(float* __restrict__ hio, const float* __restrict__ bias,
                              const float* __restrict__ g, const float* __restrict__ be,
                              const float* __restrict__ mth, const float* __restrict__ vth,
                              const float* __restrict__ resid, int n_nodes) {
  size_t idx = (size_t)blockIdx.x * blockDim.x + threadIdx.x;
  size_t total = (size_t)n_nodes * HID;
  if (idx >= total) return;
  int f = (int)(idx & (HID - 1));
  float t = hio[idx] + bias[f];
  t = (t - mth[f]) * g[f] * rsqrtf(vth[f] + BN_EPS) + be[f];
  t = (t > 0.f) ? t : (__expf(t) - 1.f);
  if (resid) t += resid[idx];
  hio[idx] = t;
}

__global__ void bn_relu_k(float* __restrict__ z, const float* __restrict__ g,
                          const float* __restrict__ be, const float* __restrict__ mth,
                          const float* __restrict__ vth, int n_nodes) {
  size_t idx = (size_t)blockIdx.x * blockDim.x + threadIdx.x;
  size_t total = (size_t)n_nodes * HHID;
  if (idx >= total) return;
  int f = (int)(idx & (HHID - 1));
  float t = (z[idx] - mth[f]) * g[f] * rsqrtf(vth[f] + BN_EPS) + be[f];
  z[idx] = (t > 0.f) ? t : 0.f;
}

// ---- exact top-k selection: 64K-bucket histogram + threshold + compact ----
__global__ void score_hist_k(const float* __restrict__ score, unsigned* __restrict__ hist,
                             int total) {
  int e = blockIdx.x * blockDim.x + threadIdx.x;
  if (e >= total) return;
  atomicAdd(&hist[orderKey(score[e]) >> 16], 1u);
}

__global__ void find_thresh_k(const unsigned* __restrict__ hist, int k,
                              int* __restrict__ thr) {
  __shared__ unsigned csum[1024];
  int t = threadIdx.x;
  unsigned s = 0;
  for (int i = 0; i < 64; ++i) s += hist[t * 64 + i];
  csum[t] = s;
  __syncthreads();
  if (t == 0) {
    unsigned cum = 0; int T = 0; unsigned above = 0;
    for (int c = 1023; c >= 0; --c) {
      if (cum + csum[c] >= (unsigned)k) {
        unsigned cc = cum;
        for (int b = c * 64 + 63; b >= c * 64; --b) {
          unsigned hb = hist[b];
          if (cc + hb >= (unsigned)k) { T = b; above = cc; break; }
          cc += hb;
        }
        break;
      }
      cum += csum[c];
    }
    thr[0] = T; thr[1] = (int)above; thr[2] = k - (int)above;
  }
}

__global__ void compact_k(const int* __restrict__ srcA, const int* __restrict__ dstA,
                          int ne, int n_nodes, const float* __restrict__ score,
                          const int* __restrict__ thr, int* __restrict__ osrc,
                          int* __restrict__ odst, int* __restrict__ counters) {
  int e = blockIdx.x * blockDim.x + threadIdx.x;
  int total = ne + n_nodes;
  if (e >= total) return;
  int key16 = (int)(orderKey(score[e]) >> 16);
  int T = thr[0], base = thr[1], r = thr[2];
  int s = (e < ne) ? srcA[e] : (e - ne);
  int d = (e < ne) ? dstA[e] : (e - ne);
  if (key16 > T) {
    int p = atomicAdd(&counters[0], 1);
    osrc[p] = s; odst[p] = d;
  } else if (key16 == T) {
    int p = atomicAdd(&counters[1], 1);
    if (p < r) { osrc[base + p] = s; odst[base + p] = d; }
  }
}

// ---- fused classifier head: logits = z@Wc2 + bc2; log_softmax per row ----
__global__ void logits_lsm_k(const float* __restrict__ z, const float* __restrict__ Wc2,
                             const float* __restrict__ bc2, float* __restrict__ outp,
                             int n_nodes) {
  int row = blockIdx.x * (blockDim.x >> 5) + (threadIdx.x >> 5);
  int lane = threadIdx.x & 31;
  if (row >= n_nodes) return;
  const float* zr = z + (size_t)row * HHID;
  float v0, v1 = -3.0e38f;
  {
    float acc = bc2[lane];
    for (int k2 = 0; k2 < HHID; ++k2) acc += zr[k2] * Wc2[k2 * OUT_C + lane];
    v0 = acc;
  }
  if (lane < OUT_C - 32) {
    float acc = bc2[lane + 32];
    for (int k2 = 0; k2 < HHID; ++k2) acc += zr[k2] * Wc2[k2 * OUT_C + lane + 32];
    v1 = acc;
  }
  float mxv = fmaxf(v0, v1);
  for (int o = 16; o > 0; o >>= 1) mxv = fmaxf(mxv, __shfl_xor(mxv, o, 32));
  float se = __expf(v0 - mxv) + ((lane < OUT_C - 32) ? __expf(v1 - mxv) : 0.f);
  for (int o = 16; o > 0; o >>= 1) se += __shfl_xor(se, o, 32);
  float lse = mxv + __logf(se);
  outp[(size_t)row * OUT_C + lane] = v0 - lse;
  if (lane < OUT_C - 32) outp[(size_t)row * OUT_C + lane + 32] = v1 - lse;
}

// ======================================================================
extern "C" void kernel_launch(void* const* d_in, const int* in_sizes, int n_in,
                              void* d_out, int out_size, void* d_ws, size_t ws_size,
                              hipStream_t stream) {
  (void)n_in; (void)out_size; (void)ws_size;
  const float* x     = (const float*)d_in[0];
  const int*   ei    = (const int*)d_in[1];
  const float* Wr    = (const float*)d_in[2];
  const float* br    = (const float*)d_in[3];
  const float* W1    = (const float*)d_in[4];
  const float* asrc1 = (const float*)d_in[5];
  const float* adst1 = (const float*)d_in[6];
  const float* b1    = (const float*)d_in[7];
  const float* g1    = (const float*)d_in[8];
  const float* be1   = (const float*)d_in[9];
  const float* m1    = (const float*)d_in[10];
  const float* v1    = (const float*)d_in[11];
  const float* W2    = (const float*)d_in[12];
  const float* asrc2 = (const float*)d_in[13];
  const float* adst2 = (const float*)d_in[14];
  const float* b2    = (const float*)d_in[15];
  const float* g2    = (const float*)d_in[16];
  const float* be2   = (const float*)d_in[17];
  const float* m2    = (const float*)d_in[18];
  const float* v2    = (const float*)d_in[19];
  const float* Wc1   = (const float*)d_in[20];
  const float* bc1   = (const float*)d_in[21];
  const float* gc    = (const float*)d_in[22];
  const float* bec   = (const float*)d_in[23];
  const float* mc    = (const float*)d_in[24];
  const float* vc    = (const float*)d_in[25];
  const float* Wc2   = (const float*)d_in[26];
  const float* bc2   = (const float*)d_in[27];

  const int n_nodes = in_sizes[0] / IN_F;
  const int n_edges = in_sizes[1] / 2;
  const int tot1 = n_edges + n_nodes;
  int kkeep = (int)((double)tot1 * 0.3);   // replicates Python int() truncation
  if (kkeep < 1) kkeep = 1;
  if (kkeep > tot1) kkeep = tot1;

  // ---- carve workspace ----
  char* wsb = (char*)d_ws;
  size_t off = 0;
  auto carve = [&](size_t bytes) -> void* {
    void* p = wsb + off;
    off += (bytes + 255) & ~(size_t)255;
    return p;
  };
  float*    xproj = (float*)carve((size_t)n_nodes * HID * 4);
  float*    xp    = (float*)carve((size_t)n_nodes * HID * 4);   // also reused as z
  float*    hbuf  = (float*)carve((size_t)n_nodes * HID * 4);
  float*    h2buf = (float*)carve((size_t)n_nodes * HID * 4);
  float*    as_   = (float*)carve((size_t)n_nodes * NH * 4);
  float*    ad_   = (float*)carve((size_t)n_nodes * NH * 4);
  unsigned* mx    = (unsigned*)carve((size_t)n_nodes * NH * 4);
  float*    den   = (float*)carve((size_t)n_nodes * NH * 4);
  float*    elog  = (float*)carve((size_t)tot1 * NH * 4);
  float*    score = (float*)carve((size_t)tot1 * 4);
  unsigned* hist  = (unsigned*)carve(65536 * 4);
  int*      thr   = (int*)carve(64);
  int*      cnts  = (int*)carve(64);
  int*      e2src = (int*)carve((size_t)kkeep * 4);
  int*      e2dst = (int*)carve((size_t)kkeep * 4);
  __bf16*   bpack = (__bf16*)carve((size_t)(HID / 32) * (HID / 16) * 32 * 16 * 2);

  const int TB = 256;
  auto g1d = [](long long n) { return (unsigned)((n + 255) / 256); };
  float* zbuf = xp;

  // ---- 1) x_proj = x @ Wr + br ----
  {
    int tot = (IN_F / 32) * (HID / 16) * 32;
    pack_b_k<<<g1d(tot), TB, 0, stream>>>(Wr, bpack, IN_F, HID);
    dim3 grid((((n_nodes + 15) / 16) + 7) / 8, HID / 64);
    gemm_wmma_k<<<grid, TB, 0, stream>>>(x, bpack, br, xproj, n_nodes, IN_F, HID);
  }

  // ---- 2) GAT layer 1 ----
  {
    int tot = (HID / 32) * (HID / 16) * 32;
    pack_b_k<<<g1d(tot), TB, 0, stream>>>(W1, bpack, HID, HID);
    dim3 grid((((n_nodes + 15) / 16) + 7) / 8, HID / 64);
    gemm_wmma_k<<<grid, TB, 0, stream>>>(xproj, bpack, nullptr, xp, n_nodes, HID, HID);
    attn_coeff_k<<<g1d((long long)n_nodes * NH), TB, 0, stream>>>(xp, asrc1, adst1, as_, ad_, n_nodes);

    hipMemsetAsync(mx, 0, (size_t)n_nodes * NH * 4, stream);
    hipMemsetAsync(den, 0, (size_t)n_nodes * NH * 4, stream);
    hipMemsetAsync(hbuf, 0, (size_t)n_nodes * HID * 4, stream);

    const int* s1 = ei;
    const int* d1 = ei + n_edges;
    long long th = (long long)tot1 * NH;
    edge_logits_k<<<g1d(th), TB, 0, stream>>>(s1, d1, n_edges, n_nodes, as_, ad_, elog, mx);
    edge_exp_k<<<g1d(th), TB, 0, stream>>>(d1, n_edges, n_nodes, elog, mx, den);
    edge_alpha_k<<<g1d(th), TB, 0, stream>>>(d1, n_edges, n_nodes, elog, den);
    edge_score_k<<<g1d(tot1), TB, 0, stream>>>(elog, score, tot1);
    aggregate_k<<<g1d((long long)tot1 * HID), TB, 0, stream>>>(s1, d1, n_edges, n_nodes, xp, elog, hbuf);
    bias_bn_elu_k<<<g1d((long long)n_nodes * HID), TB, 0, stream>>>(hbuf, b1, g1, be1, m1, v1, nullptr, n_nodes);

    // ---- 3) top-k edge pruning ----
    hipMemsetAsync(hist, 0, 65536 * 4, stream);
    hipMemsetAsync(cnts, 0, 64, stream);
    score_hist_k<<<g1d(tot1), TB, 0, stream>>>(score, hist, tot1);
    find_thresh_k<<<1, 1024, 0, stream>>>(hist, kkeep, thr);
    compact_k<<<g1d(tot1), TB, 0, stream>>>(s1, d1, n_edges, n_nodes, score, thr, e2src, e2dst, cnts);
  }

  // ---- 4) GAT layer 2 (pruned edges + fresh self loops) ----
  {
    int tot = (HID / 32) * (HID / 16) * 32;
    pack_b_k<<<g1d(tot), TB, 0, stream>>>(W2, bpack, HID, HID);
    dim3 grid((((n_nodes + 15) / 16) + 7) / 8, HID / 64);
    gemm_wmma_k<<<grid, TB, 0, stream>>>(hbuf, bpack, nullptr, xp, n_nodes, HID, HID);
    attn_coeff_k<<<g1d((long long)n_nodes * NH), TB, 0, stream>>>(xp, asrc2, adst2, as_, ad_, n_nodes);

    hipMemsetAsync(mx, 0, (size_t)n_nodes * NH * 4, stream);
    hipMemsetAsync(den, 0, (size_t)n_nodes * NH * 4, stream);
    hipMemsetAsync(h2buf, 0, (size_t)n_nodes * HID * 4, stream);

    int tot2 = kkeep + n_nodes;
    long long th = (long long)tot2 * NH;
    edge_logits_k<<<g1d(th), TB, 0, stream>>>(e2src, e2dst, kkeep, n_nodes, as_, ad_, elog, mx);
    edge_exp_k<<<g1d(th), TB, 0, stream>>>(e2dst, kkeep, n_nodes, elog, mx, den);
    edge_alpha_k<<<g1d(th), TB, 0, stream>>>(e2dst, kkeep, n_nodes, elog, den);
    aggregate_k<<<g1d((long long)tot2 * HID), TB, 0, stream>>>(e2src, e2dst, kkeep, n_nodes, xp, elog, h2buf);
    bias_bn_elu_k<<<g1d((long long)n_nodes * HID), TB, 0, stream>>>(h2buf, b2, g2, be2, m2, v2, xproj, n_nodes);
  }

  // ---- 5) classifier: z = relu(bn(h2@Wc1 + bc1)); log_softmax(z@Wc2 + bc2) ----
  {
    int tot = (HID / 32) * (HHID / 16) * 32;
    pack_b_k<<<g1d(tot), TB, 0, stream>>>(Wc1, bpack, HID, HHID);
    dim3 grid((((n_nodes + 15) / 16) + 7) / 8, HHID / 64);
    gemm_wmma_k<<<grid, TB, 0, stream>>>(h2buf, bpack, bc1, zbuf, n_nodes, HID, HHID);
    bn_relu_k<<<g1d((long long)n_nodes * HHID), TB, 0, stream>>>(zbuf, gc, bec, mc, vc, n_nodes);
    logits_lsm_k<<<(n_nodes + 7) / 8, TB, 0, stream>>>(zbuf, Wc2, bc2, (float*)d_out, n_nodes);
  }
}
